// DTWCalculator_57741540327822
// MI455X (gfx1250) — matrix operand order repaired and checked by reading, not compiled, not run on hardware
//
#include <hip/hip_runtime.h>
#include <hip/hip_bf16.h>
#include <float.h>

typedef __attribute__((ext_vector_type(16))) _Float16 v16h;
typedef __attribute__((ext_vector_type(8)))  _Float16 v8h;
typedef __attribute__((ext_vector_type(8)))  float    v8f;
typedef unsigned int u32;
typedef __attribute__((ext_vector_type(4))) u32 u32x4;
typedef __attribute__((ext_vector_type(8))) u32 u32x8;

#define D_DIM 1024
#define N_ROWS 4096
#define N_COLS 4096
#define BM 128
#define BN 64
#define BK 32
#define LDA (BK + 8)   // padded stride in halves: 32 halves data + 8 halves pad = 80B
#define LDB (BK + 8)

// ---------------------------------------------------------------------------
// Kernel 1: row L2-normalize f32 -> f16 (one block per row, 256 threads)
// ---------------------------------------------------------------------------
__global__ __launch_bounds__(256) void normalize_rows_f16(
    const float* __restrict__ src, _Float16* __restrict__ dst) {
  __shared__ float wsum[8];
  const int row  = blockIdx.x;
  const int tid  = threadIdx.x;
  const int lane = tid & 31;
  const int wave = tid >> 5;
  const float* r = src + (size_t)row * D_DIM;

  float s = 0.f;
  #pragma unroll
  for (int k = 0; k < 4; ++k) {
    float x = r[tid + k * 256];
    s += x * x;
  }
  #pragma unroll
  for (int off = 16; off; off >>= 1) s += __shfl_xor(s, off, 32);
  if (lane == 0) wsum[wave] = s;
  __syncthreads();
  if (tid == 0) {
    float t = 0.f;
    #pragma unroll
    for (int w = 0; w < 8; ++w) t += wsum[w];
    wsum[0] = 1.0f / (sqrtf(t) + 1e-8f);
  }
  __syncthreads();
  const float scale = wsum[0];
  _Float16* o = dst + (size_t)row * D_DIM;
  #pragma unroll
  for (int k = 0; k < 4; ++k)
    o[tid + k * 256] = (_Float16)(r[tid + k * 256] * scale);
}

// ---------------------------------------------------------------------------
// TDM: 2D tile load Global -> LDS with row padding (pad 4 dwords every 16
// dwords => LDS row stride 40 halves, matching As/Bs layout).
// D# group0: count=1 | lds_addr | global_addr[56:0] | type=2
// D# group1: data_size=2B, pad_enable, pad_interval=16dw, pad_amount=4dw,
//            tensor_dim0/1, tile_dim0/1, tensor_dim0_stride
// ---------------------------------------------------------------------------
__device__ __forceinline__ void tdm_load_tile_2d(
    u32 lds_addr, const void* gptr,
    u32 tile_d0, u32 tile_d1, u32 tensor_d0, u32 tensor_d1, u32 stride_e) {
  unsigned long long ga = (unsigned long long)(uintptr_t)gptr;
  u32x4 g0;
  g0[0] = 1u;                                  // count=1 (valid), user mode
  g0[1] = lds_addr;                            // LDS byte address
  g0[2] = (u32)ga;                             // global addr [31:0]
  g0[3] = (u32)(ga >> 32) | (2u << 30);        // global addr [56:32] | type=2
  u32x8 g1;
  g1[0] = (1u << 16)                           // data_size = 2 bytes
        | (1u << 20)                           // pad_enable
        | (3u << 22)                           // pad_interval: 16 dwords
        | (3u << 25);                          // pad_amount: 4 dwords
  g1[1] = (tensor_d0 & 0xFFFFu) << 16;         // tensor_dim0 [15:0]
  g1[2] = (tensor_d0 >> 16) | ((tensor_d1 & 0xFFFFu) << 16);
  g1[3] = (tensor_d1 >> 16) | (tile_d0 << 16); // tile_dim0
  g1[4] = tile_d1;                             // tile_dim1 (tile_dim2 = 0)
  g1[5] = stride_e;                            // tensor_dim0_stride [31:0]
  g1[6] = 0u;
  g1[7] = 0u;
  asm volatile("tensor_load_to_lds %0, %1" :: "s"(g0), "s"(g1) : "memory");
}

// ---------------------------------------------------------------------------
// Kernel 2: cost[i][j] = 1 - A_hat[i,:].B_hat[j,:] via v_wmma_f32_16x16x32_f16
// 256 threads (8 waves); block tile 128x64; wave tile 32x32 (2x2 WMMA tiles).
// Double-buffered LDS tiles filled by TDM, overlapped with WMMA compute.
// ---------------------------------------------------------------------------
__global__ __launch_bounds__(256) void cost_gemm_wmma(
    const _Float16* __restrict__ A, const _Float16* __restrict__ B,
    float* __restrict__ cost) {
  __shared__ _Float16 As[2][BM][LDA];
  __shared__ _Float16 Bs[2][BN][LDB];

  const int tid     = threadIdx.x;
  const int lane    = tid & 31;
  const int wave    = tid >> 5;
  const int waveM   = wave & 3;        // 0..3 -> M offset 0..96
  const int waveN   = wave >> 2;       // 0..1 -> N offset 0/32
  const int rowBase = blockIdx.y * BM;
  const int colBase = blockIdx.x * BN;
  const int halfSel = (lane >> 4) & 1; // 0 for lanes 0-15, 1 for 16-31
  const int l15     = lane & 15;

  v8f acc[2][2] = {};

  auto issue_tiles = [&](int buf, int k0) {
    if (wave == 0) {
      tdm_load_tile_2d((u32)(uintptr_t)(void*)&As[buf][0][0],
                       A + (size_t)rowBase * D_DIM + k0,
                       BK, BM, D_DIM, N_ROWS, D_DIM);
    } else if (wave == 1) {
      tdm_load_tile_2d((u32)(uintptr_t)(void*)&Bs[buf][0][0],
                       B + (size_t)colBase * D_DIM + k0,
                       BK, BN, D_DIM, N_COLS, D_DIM);
    }
  };

  // Prologue: DMA tile 0 into buffer 0. TENSORcnt is 0 in non-issuing waves,
  // so waiting unconditionally avoids any exec-mask divergence.
  issue_tiles(0, 0);
  __builtin_amdgcn_s_wait_tensorcnt(0);
  __syncthreads();

  int buf = 0;
  for (int k0 = 0; k0 < D_DIM; k0 += BK, buf ^= 1) {
    // Overlap: DMA next tile into the other buffer while computing this one
    if (k0 + BK < D_DIM) issue_tiles(buf ^ 1, k0 + BK);

    // A frag (16x32 f16): lanes 0-15 hold K 0..7/16..23, lanes 16-31 K 8..15/24..31
    v16h afrag[2], bfrag[2];
    #pragma unroll
    for (int mi = 0; mi < 2; ++mi) {
      int r   = waveM * 32 + mi * 16 + l15;
      int klo = halfSel * 8;
      v8h lo = *(const v8h*)&As[buf][r][klo];
      v8h hi = *(const v8h*)&As[buf][r][klo + 16];
      #pragma unroll
      for (int e = 0; e < 8; ++e) { afrag[mi][e] = lo[e]; afrag[mi][8 + e] = hi[e]; }
    }
    // B frag (32x16 f16): lanes 0-15 hold K 0..7/16..23, lanes 16-31 K 8..15/24..31; N = lane&15
    #pragma unroll
    for (int ni = 0; ni < 2; ++ni) {
      int r  = waveN * 32 + ni * 16 + l15;
      int kb = halfSel * 8;
      v8h lo = *(const v8h*)&Bs[buf][r][kb];
      v8h hi = *(const v8h*)&Bs[buf][r][kb + 16];
      #pragma unroll
      for (int e = 0; e < 8; ++e) { bfrag[ni][e] = lo[e]; bfrag[ni][8 + e] = hi[e]; }
    }
    #pragma unroll
    for (int mi = 0; mi < 2; ++mi)
      #pragma unroll
      for (int ni = 0; ni < 2; ++ni)
        acc[mi][ni] = __builtin_amdgcn_wmma_f32_16x16x32_f16(
            false, afrag[mi], false, bfrag[ni], (short)0, acc[mi][ni],
            false, false);

    // Next tile's DMA must land, and all waves must be done reading this
    // buffer, before the next iteration flips buffers.
    __builtin_amdgcn_s_wait_tensorcnt(0);
    __syncthreads();
  }

  // C/D layout: VGPR v -> M = v + 8*halfSel ; N = lane&15
  // N_COLS is compile-time so row addressing lowers to shifts.
  #pragma unroll
  for (int mi = 0; mi < 2; ++mi)
    #pragma unroll
    for (int ni = 0; ni < 2; ++ni) {
      int row = rowBase + waveM * 32 + mi * 16 + halfSel * 8;
      int col = colBase + waveN * 32 + ni * 16 + l15;
      #pragma unroll
      for (int v = 0; v < 8; ++v)
        cost[((size_t)(row + v) << 12) + col] = 1.0f - acc[mi][ni][v];
    }
}

// ---------------------------------------------------------------------------
// Kernel 3: DTW DP, single persistent block of 1024 threads (32 waves).
// Row recurrence solved as: S = cumsum(c); D = cummin(a - S) + S.
// ---------------------------------------------------------------------------
template<bool IS_MIN>
__device__ __forceinline__ void block_scan4(float v[4], float* wag, int tid) {
  const float ident = IS_MIN ? FLT_MAX : 0.0f;
  auto op = [](float a, float b) { return IS_MIN ? fminf(a, b) : a + b; };
  v[1] = op(v[1], v[0]); v[2] = op(v[2], v[1]); v[3] = op(v[3], v[2]);
  const int lane = tid & 31, wave = tid >> 5;
  float ws = v[3];
  #pragma unroll
  for (int off = 1; off < 32; off <<= 1) {
    float u = __shfl_up(ws, off, 32);
    if (lane >= off) ws = op(ws, u);
  }
  if (lane == 31) wag[wave] = ws;
  __syncthreads();
  if (wave == 0) {               // scan the 32 wave totals
    float sv = wag[lane];
    #pragma unroll
    for (int off = 1; off < 32; off <<= 1) {
      float u = __shfl_up(sv, off, 32);
      if (lane >= off) sv = op(sv, u);
    }
    wag[lane] = sv;
  }
  __syncthreads();
  float waveExcl = wave ? wag[wave - 1] : ident;
  float thrExcl  = __shfl_up(ws, 1, 32);
  if (lane == 0) thrExcl = ident;
  float pre = op(waveExcl, thrExcl);
  v[0] = op(v[0], pre); v[1] = op(v[1], pre);
  v[2] = op(v[2], pre); v[3] = op(v[3], pre);
  __syncthreads();               // protect wag for the next scan
}

__global__ __launch_bounds__(1024) void dtw_kernel(
    const float* __restrict__ cost, float* __restrict__ out) {
  __shared__ float prevRow[N_COLS];
  __shared__ float wag[32];
  const int tid = threadIdx.x;
  const int j0  = tid * 4;

  // Row 0: plain cumsum
  float c[4];
  #pragma unroll
  for (int k = 0; k < 4; ++k) c[k] = cost[j0 + k];
  block_scan4<false>(c, wag, tid);
  #pragma unroll
  for (int k = 0; k < 4; ++k) prevRow[j0 + k] = c[k];
  __syncthreads();

  for (int i = 1; i < N_ROWS; ++i) {
    const float* crow = cost + ((size_t)i << 12);
    #pragma unroll
    for (int k = 0; k < 4; ++k) c[k] = crow[j0 + k];
    if (i + 1 < N_ROWS)
      __builtin_prefetch(cost + ((size_t)(i + 1) << 12) + j0, 0, 1);

    float a[4];
    #pragma unroll
    for (int k = 0; k < 4; ++k) {
      int j = j0 + k;
      float pd = (j > 0) ? prevRow[j - 1] : FLT_MAX;
      a[k] = c[k] + fminf(prevRow[j], pd);
    }
    float S[4] = {c[0], c[1], c[2], c[3]};
    block_scan4<false>(S, wag, tid);             // S = cumsum(c)
    float t[4];
    #pragma unroll
    for (int k = 0; k < 4; ++k) t[k] = a[k] - S[k];
    block_scan4<true>(t, wag, tid);              // cummin(a - S)
    #pragma unroll
    for (int k = 0; k < 4; ++k) prevRow[j0 + k] = t[k] + S[k];
    __syncthreads();
  }
  if (tid == 1023) out[0] = 1.0f / (1.0f + prevRow[N_COLS - 1]);
}

// ---------------------------------------------------------------------------
extern "C" void kernel_launch(void* const* d_in, const int* in_sizes, int n_in,
                              void* d_out, int out_size, void* d_ws, size_t ws_size,
                              hipStream_t stream) {
  const float* t1 = (const float*)d_in[0];
  const float* t2 = (const float*)d_in[1];
  const int D  = D_DIM;
  const int N1 = in_sizes[0] / D;   // 4096
  const int N2 = in_sizes[1] / D;   // 4096

  char* ws = (char*)d_ws;
  _Float16* Af  = (_Float16*)ws;
  _Float16* Bf  = (_Float16*)(ws + (size_t)N1 * D * sizeof(_Float16));
  float*    cst = (float*)(ws + (size_t)(N1 + N2) * D * sizeof(_Float16));

  normalize_rows_f16<<<N1, 256, 0, stream>>>(t1, Af);
  normalize_rows_f16<<<N2, 256, 0, stream>>>(t2, Bf);

  dim3 grid(N2 / BN, N1 / BM);
  cost_gemm_wmma<<<grid, 256, 0, stream>>>(Af, Bf, cst);

  dtw_kernel<<<1, 1024, 0, stream>>>(cst, (float*)d_out);
}
